// MultiHeadLatentAttention_36258113913353
// MI455X (gfx1250) — matrix operand order repaired and checked by reading, not compiled, not run on hardware
//
#include <hip/hip_runtime.h>
#include <hip/hip_bf16.h>

typedef __attribute__((ext_vector_type(8)))  _Float16 v8h;
typedef __attribute__((ext_vector_type(16))) _Float16 v16h;
typedef __attribute__((ext_vector_type(8)))  float    v8f;
typedef __attribute__((ext_vector_type(4)))  unsigned int v4u;
typedef __attribute__((ext_vector_type(8)))  int v8i;
typedef __attribute__((ext_vector_type(4)))  int v4i;

#define DIM    1024
#define NHEADS 16
#define HDIM   64
#define LATENT 128
#define QRANK  256
#define BATCH  4
#define SEQ    2048
#define MTOT   (BATCH * SEQ)   // 8192 rows

static __device__ __forceinline__ v16h cat8(v8h lo, v8h hi) {
  return __builtin_shufflevector(lo, hi, 0,1,2,3,4,5,6,7,8,9,10,11,12,13,14,15);
}
static __device__ __forceinline__ v8f wmma16(v16h a, v16h b, v8f c) {
  return __builtin_amdgcn_wmma_f32_16x16x32_f16(false, a, false, b, (short)0, c, false, false);
}

// ---- Tensor Data Mover: 2D tile (16-bit elements) global -> LDS -------------
// D# per CDNA5 ISA  (group0 128b, group1 256b; groups 2/3
// zero for a 2D tensor).  lds_addr = low 32 bits of the generic pointer
// Issue from one thread; EXEC is
// ignored by TDM so a single-lane wave performs the full tile DMA.
// This toolchain uses the 6-arg builtin form:
//   (uint32x4 g0, int32x8 g1, int32x4 g2, int32x4 g3, int32x8 pad, i32 cpol)
static __device__ __forceinline__ void tdm_load_2d_f16(uint32_t lds_addr,
                                                       const void* gtile,
                                                       uint32_t tile_w,   // elements along dim0
                                                       uint32_t tile_h,   // rows
                                                       uint64_t row_stride_elems) {
  const uint64_t ga = (uint64_t)(uintptr_t)gtile;
  v4u g0;
  g0[0] = 1u;                                              // count=1 (valid user D#)
  g0[1] = lds_addr;                                        // bits 63:32
  g0[2] = (uint32_t)ga;                                    // bits 95:64
  g0[3] = ((uint32_t)(ga >> 32) & 0x01FFFFFFu) | (2u << 30); // addr[56:32] | type=2
  uint32_t w[8];
#pragma unroll
  for (int i = 0; i < 8; ++i) w[i] = 0u;
  w[0] = (1u << 16);                                       // data_size=1 (2 bytes), mask=0
  w[1] = (tile_w & 0xFFFFu) << 16;                         // tensor_dim0[15:0]  @63:48
  w[2] = (tile_w >> 16) | ((tile_h & 0xFFFFu) << 16);      // tensor_dim0[31:16] | tensor_dim1[15:0]
  w[3] = (tile_h >> 16) | (tile_w << 16);                  // tensor_dim1[31:16] | tile_dim0 @127:112
  w[4] = tile_h;                                           // tile_dim1 @143:128 (tile_dim2=0)
  w[5] = (uint32_t)row_stride_elems;                       // tensor_dim0_stride[31:0] @191:160
  w[6] = (uint32_t)((row_stride_elems >> 32) & 0xFFFFu);   // stride[47:32]; dim1_stride=0
  v8i g1;
#pragma unroll
  for (int i = 0; i < 8; ++i) g1[i] = (int)w[i];
  v4i z4;
  z4[0] = 0; z4[1] = 0; z4[2] = 0; z4[3] = 0;
  v8i z8;
#pragma unroll
  for (int i = 0; i < 8; ++i) z8[i] = 0;
  __builtin_amdgcn_tensor_load_to_lds(g0, g1, z4, z4, z8, 0);
}

__global__ __launch_bounds__(256) void cvt_f32_f16(const float* __restrict__ in,
                                                   _Float16* __restrict__ out, int n) {
  int i = blockIdx.x * 256 + threadIdx.x;
  if (i < n) out[i] = (_Float16)in[i];
}

// Row-major C[M,N] = A[M,K] @ B[K,N] + bias[N].
// BM=128 BN=128 BK=32, 8 waves (4 row x 2 col), wave tile 32x64 -> 8 WMMA/iter.
// A tile loaded by the Tensor Data Mover; B tile transposed into LDS by VALU.
template <bool OUT_F32>
__global__ __launch_bounds__(256) void gemm_f16(const _Float16* __restrict__ A,
                                                const _Float16* __restrict__ B,
                                                const float* __restrict__ bias,
                                                void* __restrict__ Cp,
                                                int M, int N, int K) {
  __shared__ __align__(16) _Float16 Al[128 * 32];    // A tile row-major (TDM dest)
  __shared__ __align__(16) _Float16 Btl[128 * 32];   // B tile transposed: Btl[n][k]

  const int tid  = threadIdx.x;
  const int lane = tid & 31;
  const int wave = tid >> 5;
  const int wr   = wave & 3;     // 0..3 : wave row (32 rows each)
  const int wc   = wave >> 2;    // 0..1 : wave col (64 cols each)
  const int lr   = lane & 15;
  const int lh   = lane >> 4;
  const int mBase = blockIdx.y * 128;
  const int nBase = blockIdx.x * 128;
  const uint32_t ldsA = (uint32_t)(uintptr_t)&Al[0];

  v8f acc[2][4];
#pragma unroll
  for (int i = 0; i < 2; ++i)
#pragma unroll
    for (int j = 0; j < 4; ++j)
#pragma unroll
      for (int r = 0; r < 8; ++r) acc[i][j][r] = 0.0f;

  for (int k0 = 0; k0 < K; k0 += 32) {
    // ---- async A tile via TDM (issued once per workgroup) ----
    if (tid == 0)
      tdm_load_2d_f16(ldsA, A + (size_t)mBase * K + k0, 32u, 128u, (uint64_t)K);
    // ---- B tile: global load + transpose scatter into LDS ----
#pragma unroll
    for (int c = 0; c < 2; ++c) {
      const int idx = tid * 16 + c * 8;
      const int bk  = idx >> 7;         // 0..31
      const int bn  = idx & 127;        // multiple of 8
      v8h bv = *(const v8h*)(B + (size_t)(k0 + bk) * N + nBase + bn);
#pragma unroll
      for (int j = 0; j < 8; ++j) Btl[(bn + j) * 32 + bk] = bv[j];
      if (k0 + 32 < K)
        __builtin_prefetch(B + (size_t)(k0 + 32 + bk) * N + nBase + bn, 0, 1);
    }
    if (tid == 0) __builtin_amdgcn_s_wait_tensorcnt(0);    // TENSORcnt is per-wave
    __syncthreads();                                       // publish tiles

    // ---- fragments + WMMA ----
    v16h af[2], bf[4];
#pragma unroll
    for (int mt = 0; mt < 2; ++mt) {
      const _Float16* p = &Al[(wr * 32 + mt * 16 + lr) * 32 + lh * 8];
      af[mt] = cat8(*(const v8h*)p, *(const v8h*)(p + 16));
    }
#pragma unroll
    for (int nt = 0; nt < 4; ++nt) {
      const _Float16* p = &Btl[(wc * 64 + nt * 16 + lr) * 32 + lh * 8];
      bf[nt] = cat8(*(const v8h*)p, *(const v8h*)(p + 16));
    }
#pragma unroll
    for (int mt = 0; mt < 2; ++mt)
#pragma unroll
      for (int nt = 0; nt < 4; ++nt)
        acc[mt][nt] = wmma16(af[mt], bf[nt], acc[mt][nt]);
    __syncthreads();
  }

  // ---- epilogue: bias + store (C layout: VGPR r -> row r / 8+r per half-wave) ----
  const int rofs = (lane >= 16) ? 8 : 0;
#pragma unroll
  for (int mt = 0; mt < 2; ++mt)
#pragma unroll
    for (int nt = 0; nt < 4; ++nt)
#pragma unroll
      for (int r = 0; r < 8; ++r) {
        const int row = mBase + wr * 32 + mt * 16 + r + rofs;
        const int col = nBase + wc * 64 + nt * 16 + lr;
        const float v = acc[mt][nt][r] + bias[col];
        if (OUT_F32) ((float*)Cp)[(size_t)row * N + col] = v;
        else         ((_Float16*)Cp)[(size_t)row * N + col] = (_Float16)v;
      }
}

// Flash attention, causal.  Qb[MTOT,DIM] f16, KVb[MTOT,2*DIM] f16 (K cols 0..1023, V cols 1024..2047).
// Grid: 1024 blocks = 16 qblk x 16 heads x 4 batch; 8 waves; wave owns 16 query rows.
__global__ __launch_bounds__(256) void mla_attn(const _Float16* __restrict__ Qb,
                                                const _Float16* __restrict__ KVb,
                                                _Float16* __restrict__ Ctx) {
  __shared__ __align__(16) _Float16 Kl[32 * 64];       // K block row-major [t][d] (TDM dest)
  __shared__ __align__(16) _Float16 Vt[64 * 32];       // V block transposed [d][t]
  __shared__ __align__(16) _Float16 Pl[8][16 * 32];    // per-wave P staging

  const int tid  = threadIdx.x;
  const int lane = tid & 31;
  const int wave = tid >> 5;
  const int lr   = lane & 15;
  const int lh   = lane >> 4;
  const int bid  = blockIdx.x;
  const int qblk = bid & 15;
  const int h    = (bid >> 4) & 15;
  const int bat  = bid >> 8;
  const int qbase = qblk * 128;
  const int row0  = bat * SEQ;
  const int qwave = qbase + wave * 16;
  const int qmax  = qwave + 15;
  const int rofs  = (lane >= 16) ? 8 : 0;
  const uint32_t ldsK = (uint32_t)(uintptr_t)&Kl[0];

  // Q fragments (A layout), held for the whole pass
  v16h qf[2];
  {
    const _Float16* qp = Qb + (size_t)(row0 + qwave + lr) * DIM + h * HDIM;
#pragma unroll
    for (int ks = 0; ks < 2; ++ks) {
      const _Float16* p = qp + ks * 32 + lh * 8;
      qf[ks] = cat8(*(const v8h*)p, *(const v8h*)(p + 16));
    }
  }

  v8f o[4];
  float rowM[8], rowL[8];
#pragma unroll
  for (int dt = 0; dt < 4; ++dt)
#pragma unroll
    for (int r = 0; r < 8; ++r) o[dt][r] = 0.0f;
#pragma unroll
  for (int r = 0; r < 8; ++r) { rowM[r] = -1e30f; rowL[r] = 0.0f; }

  const int kbEnd = (qbase >> 5) + 4;   // causal range in 32-key blocks
  for (int kb = 0; kb < kbEnd; ++kb) {
    // ---- K block via TDM; V block transposed by VALU ----
    if (tid == 0)
      tdm_load_2d_f16(ldsK, KVb + (size_t)(row0 + kb * 32) * (2 * DIM) + h * HDIM,
                      64u, 32u, (uint64_t)(2 * DIM));
    {
      const int t = tid >> 3;            // 0..31
      const int d = (tid & 7) * 8;       // 0..56
      const size_t gr = (size_t)(row0 + kb * 32 + t) * (2 * DIM) + h * HDIM;
      v8h vv = *(const v8h*)(KVb + gr + DIM + d);
#pragma unroll
      for (int j = 0; j < 8; ++j) Vt[(d + j) * 32 + t] = vv[j];
      if (kb + 1 < kbEnd)
        __builtin_prefetch(KVb + gr + 32 * (2 * DIM) + DIM + d, 0, 1);
    }
    if (tid == 0) __builtin_amdgcn_s_wait_tensorcnt(0);
    __syncthreads();

    if (kb * 32 <= qmax) {
      // ---- scores: S = Q @ K^T ----
      v8f s[2];
#pragma unroll
      for (int nt = 0; nt < 2; ++nt)
#pragma unroll
        for (int r = 0; r < 8; ++r) s[nt][r] = 0.0f;
#pragma unroll
      for (int nt = 0; nt < 2; ++nt)
#pragma unroll
        for (int ks = 0; ks < 2; ++ks) {
          const _Float16* p = &Kl[(nt * 16 + lr) * 64 + ks * 32 + lh * 8];
          v16h kf = cat8(*(const v8h*)p, *(const v8h*)(p + 16));
          s[nt] = wmma16(qf[ks], kf, s[nt]);
        }

      // ---- scale + causal mask ----
#pragma unroll
      for (int nt = 0; nt < 2; ++nt)
#pragma unroll
        for (int r = 0; r < 8; ++r) {
          const int q = qwave + r + rofs;
          const int t = kb * 32 + nt * 16 + lr;
          const float v = s[nt][r] * 0.125f;     // 1/sqrt(64)
          s[nt][r] = (t <= q) ? v : -1e30f;
        }

      // ---- online softmax (row scalars per VGPR index, replicated over 16 lanes) ----
      float corr[8], ps[8];
#pragma unroll
      for (int r = 0; r < 8; ++r) {
        float rm = fmaxf(s[0][r], s[1][r]);
        rm = fmaxf(rm, __shfl_xor(rm, 1, 32));
        rm = fmaxf(rm, __shfl_xor(rm, 2, 32));
        rm = fmaxf(rm, __shfl_xor(rm, 4, 32));
        rm = fmaxf(rm, __shfl_xor(rm, 8, 32));
        const float mn = fmaxf(rowM[r], rm);
        corr[r] = __expf(rowM[r] - mn);
        rowM[r] = mn;
        ps[r] = 0.0f;
      }
#pragma unroll
      for (int nt = 0; nt < 2; ++nt)
#pragma unroll
        for (int r = 0; r < 8; ++r) {
          const float p = __expf(s[nt][r] - rowM[r]);
          s[nt][r] = p;
          ps[r] += p;
        }
#pragma unroll
      for (int r = 0; r < 8; ++r) {
        ps[r] += __shfl_xor(ps[r], 1, 32);
        ps[r] += __shfl_xor(ps[r], 2, 32);
        ps[r] += __shfl_xor(ps[r], 4, 32);
        ps[r] += __shfl_xor(ps[r], 8, 32);
        rowL[r] = rowL[r] * corr[r] + ps[r];
      }
#pragma unroll
      for (int dt = 0; dt < 4; ++dt)
#pragma unroll
        for (int r = 0; r < 8; ++r) o[dt][r] *= corr[r];

      // ---- P: C layout -> LDS -> A-fragment layout ----
#pragma unroll
      for (int nt = 0; nt < 2; ++nt)
#pragma unroll
        for (int r = 0; r < 8; ++r)
          Pl[wave][(r + rofs) * 32 + nt * 16 + lr] = (_Float16)s[nt][r];
      asm volatile("s_wait_dscnt 0x0" ::: "memory");   // per-wave LDS RAW fence
      v16h pf;
      {
        const _Float16* p = &Pl[wave][lr * 32 + lh * 8];
        pf = cat8(*(const v8h*)p, *(const v8h*)(p + 16));
      }

      // ---- O += P @ V ----
#pragma unroll
      for (int dt = 0; dt < 4; ++dt) {
        const _Float16* p = &Vt[(dt * 16 + lr) * 32 + lh * 8];
        v16h vf = cat8(*(const v8h*)p, *(const v8h*)(p + 16));
        o[dt] = wmma16(pf, vf, o[dt]);
      }
    }
    __syncthreads();
  }

  // ---- normalize + write ctx (f16) ----
#pragma unroll
  for (int dt = 0; dt < 4; ++dt)
#pragma unroll
    for (int r = 0; r < 8; ++r) {
      const int grow = row0 + qwave + r + rofs;
      const int col  = h * HDIM + dt * 16 + lr;
      Ctx[(size_t)grow * DIM + col] = (_Float16)(o[dt][r] / rowL[r]);
    }
}

extern "C" void kernel_launch(void* const* d_in, const int* in_sizes, int n_in,
                              void* d_out, int out_size, void* d_ws, size_t ws_size,
                              hipStream_t stream) {
  (void)in_sizes; (void)n_in; (void)out_size; (void)ws_size;

  const float* x     = (const float*)d_in[0];
  // d_in[1] = mask (causal, computed analytically)
  const float* w_kvc = (const float*)d_in[2];
  const float* b_kvc = (const float*)d_in[3];
  const float* w_kvu = (const float*)d_in[4];
  const float* b_kvu = (const float*)d_in[5];
  const float* w_qc  = (const float*)d_in[6];
  const float* b_qc  = (const float*)d_in[7];
  const float* w_qu  = (const float*)d_in[8];
  const float* b_qu  = (const float*)d_in[9];
  const float* w_o   = (const float*)d_in[10];
  const float* b_o   = (const float*)d_in[11];
  float* out = (float*)d_out;

  _Float16* W = (_Float16*)d_ws;
  size_t off = 0;
  _Float16* xh   = W + off; off += (size_t)MTOT * DIM;
  _Float16* wkvc = W + off; off += (size_t)DIM * LATENT;
  _Float16* wkvu = W + off; off += (size_t)LATENT * 2 * DIM;
  _Float16* wqc  = W + off; off += (size_t)DIM * QRANK;
  _Float16* wqu  = W + off; off += (size_t)QRANK * DIM;
  _Float16* wo   = W + off; off += (size_t)DIM * DIM;
  _Float16* kvl  = W + off; off += (size_t)MTOT * LATENT;
  _Float16* kvu  = W + off; off += (size_t)MTOT * 2 * DIM;
  _Float16* qc   = W + off; off += (size_t)MTOT * QRANK;
  _Float16* qh   = W + off; off += (size_t)MTOT * DIM;
  _Float16* ctx  = W + off; off += (size_t)MTOT * DIM;

  auto cvt = [&](const float* src, _Float16* dst, size_t n) {
    cvt_f32_f16<<<dim3((unsigned)((n + 255) / 256)), dim3(256), 0, stream>>>(src, dst, (int)n);
  };
  cvt(x,     xh,   (size_t)MTOT * DIM);
  cvt(w_kvc, wkvc, (size_t)DIM * LATENT);
  cvt(w_kvu, wkvu, (size_t)LATENT * 2 * DIM);
  cvt(w_qc,  wqc,  (size_t)DIM * QRANK);
  cvt(w_qu,  wqu,  (size_t)QRANK * DIM);
  cvt(w_o,   wo,   (size_t)DIM * DIM);

  // kv_latent = x @ w_kvc + b_kvc            [8192,128]
  gemm_f16<false><<<dim3(LATENT / 128, MTOT / 128), 256, 0, stream>>>(xh, wkvc, b_kvc, kvl, MTOT, LATENT, DIM);
  // kv_up = kv_latent @ w_kvu + b_kvu        [8192,2048]
  gemm_f16<false><<<dim3(2 * DIM / 128, MTOT / 128), 256, 0, stream>>>(kvl, wkvu, b_kvu, kvu, MTOT, 2 * DIM, LATENT);
  // q_c = x @ w_qc + b_qc                    [8192,256]
  gemm_f16<false><<<dim3(QRANK / 128, MTOT / 128), 256, 0, stream>>>(xh, wqc, b_qc, qc, MTOT, QRANK, DIM);
  // Q = q_c @ w_qu + b_qu                    [8192,1024]
  gemm_f16<false><<<dim3(DIM / 128, MTOT / 128), 256, 0, stream>>>(qc, wqu, b_qu, qh, MTOT, DIM, QRANK);
  // attention -> ctx                         [8192,1024]
  mla_attn<<<dim3((SEQ / 128) * NHEADS * BATCH), 256, 0, stream>>>(qh, kvu, ctx);
  // out = ctx @ w_o + b_o                    [8192,1024] f32
  gemm_f16<true><<<dim3(DIM / 128, MTOT / 128), 256, 0, stream>>>(ctx, wo, b_o, out, MTOT, DIM, DIM);
}
